// Encoder_12360915878063
// MI455X (gfx1250) — matrix operand order repaired and checked by reading, not compiled, not run on hardware
//
#include <hip/hip_runtime.h>
#include <hip/hip_bf16.h>
#include <math.h>

// Mogrifier BiLSTM encoder for MI455X (gfx1250), persistent-RNN style.
//  - bf16 weights/activations, f32 accumulation via V_WMMA_F32_16X16X32_BF16
//  - 16 workgroups per direction, atomic spin barrier (s_sleep) 4x per timestep
//  - LSTM cell state c, and final max/mean pooling, live entirely in VGPRs
//  - mogrified h staged to LDS per step (ds_load path for Whh A-fragments)
//  - gate phase: both owned (m,j) pairs fused into one K-loop -> 8 independent
//    WMMAs per 32-K chunk to hide L2 latency at ~1 wave/SIMD occupancy

#define BATCH 64
#define HID   512
#define GATES (4*HID)
#define SEQT  256
#define NWG   16
#define NWAVE 4
#define BLOCK (NWAVE*32)

typedef __attribute__((ext_vector_type(8)))  __bf16 v8bf;
typedef __attribute__((ext_vector_type(16))) __bf16 v16bf;
typedef __attribute__((ext_vector_type(8)))  float  v8f;

union V16 { v16bf v; v8bf p[2]; };

__device__ __forceinline__ float bf2f(unsigned short u) {
  return __uint_as_float(((unsigned)u) << 16);
}
__device__ __forceinline__ unsigned short f2bf(float f) {
  unsigned u = __float_as_uint(f);
  u += 0x7FFFu + ((u >> 16) & 1u);   // round-to-nearest-even
  return (unsigned short)(u >> 16);
}
__device__ __forceinline__ float sigm(float x) { return 1.0f / (1.0f + __expf(-x)); }

// Fragment loaders following the CDNA5 16-bit WMMA VGPR layouts.
// A (16xK row-major): lanes 0-15 hold K {0..7,16..23}+k0, lanes 16-31 {8..15,24..31}+k0
__device__ __forceinline__ V16 load_afrag(const unsigned short* row, int k0, int sel) {
  V16 a;
  a.p[0] = *(const v8bf*)(row + k0 + sel * 8);
  a.p[1] = *(const v8bf*)(row + k0 + 16 + sel * 8);
  return a;
}
// B column n = weight row n (row-major W[N][K]): lanes 0-15 K {0..7,16..23}, lanes 16-31 K {8..15,24..31}
__device__ __forceinline__ V16 load_bfrag(const unsigned short* row, int k0, int sel) {
  V16 b;
  b.p[0] = *(const v8bf*)(row + k0 + sel * 8);
  b.p[1] = *(const v8bf*)(row + k0 + 16 + sel * 8);
  return b;
}
__device__ __forceinline__ v8f wmma_bf16(const V16& a, const V16& b, v8f c) {
  return __builtin_amdgcn_wmma_f32_16x16x32_bf16(false, a.v, false, b.v,
                                                 (short)0, c, false, false);
}

// Single-tile K-loop with explicit double-buffered fragments (loads for chunk
// i+1 issued before the WMMA for chunk i).
template <int K>
__device__ __forceinline__ v8f wmma_kloop_pf(const unsigned short* Arow,
                                             const unsigned short* Wrow,
                                             int sel, v8f acc) {
  constexpr int NI = K / 32;
  V16 a[2], b[2];
  a[0] = load_afrag(Arow, 0, sel);
  b[0] = load_bfrag(Wrow, 0, sel);
#pragma unroll
  for (int i = 0; i < NI; ++i) {
    if (i + 1 < NI) {
      a[(i + 1) & 1] = load_afrag(Arow, (i + 1) * 32, sel);
      b[(i + 1) & 1] = load_bfrag(Wrow, (i + 1) * 32, sel);
    }
    acc = wmma_bf16(a[i & 1], b[i & 1], acc);
  }
  return acc;
}

// Fused gate K-loop: two A-streams (the wave's two (m,j) pairs) x 4 gates each
// -> 8 independent WMMAs per 32-K chunk.
template <int K>
__device__ __forceinline__ void wmma_gates(
    const unsigned short* A0, const unsigned short* A1,
    const unsigned short* w00, const unsigned short* w01,
    const unsigned short* w02, const unsigned short* w03,
    const unsigned short* w10, const unsigned short* w11,
    const unsigned short* w12, const unsigned short* w13,
    int sel, v8f (&c0)[4], v8f (&c1)[4]) {
#pragma unroll
  for (int k0 = 0; k0 < K; k0 += 32) {
    V16 a0  = load_afrag(A0, k0, sel);
    V16 a1  = load_afrag(A1, k0, sel);
    V16 b00 = load_bfrag(w00, k0, sel);
    V16 b01 = load_bfrag(w01, k0, sel);
    V16 b02 = load_bfrag(w02, k0, sel);
    V16 b03 = load_bfrag(w03, k0, sel);
    V16 b10 = load_bfrag(w10, k0, sel);
    V16 b11 = load_bfrag(w11, k0, sel);
    V16 b12 = load_bfrag(w12, k0, sel);
    V16 b13 = load_bfrag(w13, k0, sel);
    c0[0] = wmma_bf16(a0, b00, c0[0]);
    c0[1] = wmma_bf16(a0, b01, c0[1]);
    c0[2] = wmma_bf16(a0, b02, c0[2]);
    c0[3] = wmma_bf16(a0, b03, c0[3]);
    c1[0] = wmma_bf16(a1, b10, c1[0]);
    c1[1] = wmma_bf16(a1, b11, c1[1]);
    c1[2] = wmma_bf16(a1, b12, c1[2]);
    c1[3] = wmma_bf16(a1, b13, c1[3]);
  }
}

// Cross-workgroup barrier for the 16 WGs of one direction (monotonic counter).
__device__ __forceinline__ void dir_barrier(unsigned* ctr, unsigned expected) {
  __syncthreads();
  __threadfence();                    // release
  if (threadIdx.x == 0) {
    atomicAdd(ctr, 1u);
    while (__atomic_load_n((volatile unsigned*)ctr, __ATOMIC_RELAXED) < expected)
      __builtin_amdgcn_s_sleep(1);
  }
  __syncthreads();
  __threadfence();                    // acquire
}

// One mogrify round: S = A @ W^T ; edst = 2*sigmoid(S) * esrc  (elementwise)
template <int N, int K>
__device__ __forceinline__ void mogrify_phase(
    const unsigned short* A, int lda,
    const unsigned short* W, int ldw,
    const unsigned short* esrc, size_t esrc_stride,
    unsigned short* edst, int edst_stride,
    int gw, int sel, int col) {
  constexpr int total = (BATCH / 16) * (N / 16);
  for (int tile = gw; tile < total; tile += NWG * NWAVE) {
    const int m  = tile & 3;
    const int nt = tile >> 2;
    const unsigned short* Arow = A + (size_t)(m * 16 + col) * lda;
    const unsigned short* Wrow = W + (size_t)(nt * 16 + col) * ldw;
    v8f acc = {0, 0, 0, 0, 0, 0, 0, 0};
    acc = wmma_kloop_pf<K>(Arow, Wrow, sel, acc);
    const int n = nt * 16 + col;
#pragma unroll
    for (int r = 0; r < 8; r++) {
      const int brow = m * 16 + r + 8 * sel;     // D layout: row = vgpr + 8*(lane>=16)
      const float g  = 2.0f * sigm(acc[r]);
      const float xv = bf2f(esrc[(size_t)brow * esrc_stride + n]);
      edst[(size_t)brow * edst_stride + n] = f2bf(g * xv);
    }
  }
}

__global__ void zero_u32_kernel(unsigned* p, int n) {
  int i = blockIdx.x * blockDim.x + threadIdx.x;
  if (i < n) p[i] = 0u;
}

__global__ void f32_to_bf16_kernel(const float* __restrict__ src,
                                   unsigned short* __restrict__ dst, int n) {
  int i = blockIdx.x * blockDim.x + threadIdx.x;
  if (i < n) dst[i] = f2bf(src[i]);
}

// One bidirectional mogrifier-LSTM layer. grid = 32 blocks (16 per direction),
// block = 128 threads (4 waves). Persistent over all T timesteps.
template <int DIN>
__global__ void __launch_bounds__(BLOCK)
mogrifier_bilstm_layer(
    const unsigned short* __restrict__ x,                    // [B,T,DIN] bf16
    const unsigned short* Wih0, const unsigned short* Whh0,
    const unsigned short* Q0,   const unsigned short* R0,  const float* b0,
    const unsigned short* Wih1, const unsigned short* Whh1,
    const unsigned short* Q1,   const unsigned short* R1,  const float* b1,
    unsigned short* h_state,   // [2][B*H] bf16
    unsigned short* h2_buf,    // [2][B*H] bf16
    unsigned short* xt_buf,    // [2][B*DIN] bf16
    unsigned* ctrs,            // [2] barrier counters (zeroed)
    unsigned short* out_bf,    // layer1: [B,T,2H] bf16, else null
    float* out_pool)           // layer2: [B,4H] f32, else null
{
  const int dir  = (int)(blockIdx.x >> 4);
  const int wg   = (int)(blockIdx.x & 15);
  const int lane = (int)(threadIdx.x & 31);
  const int wave = (int)(threadIdx.x >> 5);
  const int gw   = wg * NWAVE + wave;        // 0..63 within direction
  const int sel  = lane >> 4;
  const int col  = lane & 15;

  const unsigned short* Wih = dir ? Wih1 : Wih0;
  const unsigned short* Whh = dir ? Whh1 : Whh0;
  const unsigned short* Qw  = dir ? Q1 : Q0;
  const unsigned short* Rw  = dir ? R1 : R0;
  const float*          bia = dir ? b1 : b0;

  unsigned short* hs = h_state + (size_t)dir * BATCH * HID;
  unsigned short* h2 = h2_buf  + (size_t)dir * BATCH * HID;
  unsigned short* xt = xt_buf  + (size_t)dir * BATCH * DIN;
  unsigned*      ctr = ctrs + dir;

  // Fixed ownership for the gate phase: 128 (m,jt) pairs over 64 waves.
  int   mq[2], jq[2];
  float bj[2][4];
#pragma unroll
  for (int q = 0; q < 2; q++) {
    int p = q * (NWG * NWAVE) + gw;
    mq[q] = p >> 5;   // batch tile 0..3
    jq[q] = p & 31;   // j tile 0..31
    int j = jq[q] * 16 + col;
#pragma unroll
    for (int g = 0; g < 4; g++) bj[q][g] = bia[g * HID + j];
  }

  float creg[2][8], mreg[2][8], sreg[2][8];
#pragma unroll
  for (int q = 0; q < 2; q++)
#pragma unroll
    for (int r = 0; r < 8; r++) { creg[q][r] = 0.f; mreg[q][r] = -1e30f; sreg[q][r] = 0.f; }

  __shared__ __attribute__((aligned(16))) unsigned short s_h2[BATCH * HID]; // 64 KB

  // Time-invariant gate-phase row pointers.
  const unsigned short* ax0 = xt + (size_t)(mq[0] * 16 + col) * DIN;
  const unsigned short* ax1 = xt + (size_t)(mq[1] * 16 + col) * DIN;
  const unsigned short* ah0 = s_h2 + (size_t)(mq[0] * 16 + col) * HID;
  const unsigned short* ah1 = s_h2 + (size_t)(mq[1] * 16 + col) * HID;
  const unsigned short* wi[2][4];
  const unsigned short* wh[2][4];
#pragma unroll
  for (int q = 0; q < 2; q++)
#pragma unroll
    for (int g = 0; g < 4; g++) {
      wi[q][g] = Wih + (size_t)(g * HID + jq[q] * 16 + col) * DIN;
      wh[q][g] = Whh + (size_t)(g * HID + jq[q] * 16 + col) * HID;
    }

  unsigned epoch = 0;
  for (int s = 0; s < SEQT; ++s) {
    const int t = dir ? (SEQT - 1 - s) : s;

    // r0: xt = 2*sigmoid(h @ Q^T) * x[:,t,:]
    mogrify_phase<DIN, HID>(hs, HID, Qw, HID,
                            x + (size_t)t * DIN, (size_t)SEQT * DIN, xt, DIN,
                            gw, sel, col);
    ++epoch; dir_barrier(ctr, epoch * NWG);

    // r1: h2 = 2*sigmoid(xt @ R^T) * h
    mogrify_phase<HID, DIN>(xt, DIN, Rw, DIN,
                            hs, HID, h2, HID, gw, sel, col);
    ++epoch; dir_barrier(ctr, epoch * NWG);

    // r2: xt = 2*sigmoid(h2 @ Q^T) * xt   (in place, one owner per element)
    mogrify_phase<DIN, HID>(h2, HID, Qw, HID,
                            xt, DIN, xt, DIN, gw, sel, col);
    ++epoch; dir_barrier(ctr, epoch * NWG);

    // Stage mogrified h into LDS for the Whh K-loop.
    {
      const uint4* s4 = (const uint4*)h2;
      uint4*       d4 = (uint4*)s_h2;
      for (int i = (int)threadIdx.x; i < (BATCH * HID) / 8; i += BLOCK) d4[i] = s4[i];
      __syncthreads();
    }

    // gates = xt @ Wih^T + h2 @ Whh^T + b ; LSTM cell update (c in VGPRs)
    v8f z = {0, 0, 0, 0, 0, 0, 0, 0};
    v8f acc[2][4] = {{z, z, z, z}, {z, z, z, z}};
    wmma_gates<DIN>(ax0, ax1,
                    wi[0][0], wi[0][1], wi[0][2], wi[0][3],
                    wi[1][0], wi[1][1], wi[1][2], wi[1][3],
                    sel, acc[0], acc[1]);
    wmma_gates<HID>(ah0, ah1,
                    wh[0][0], wh[0][1], wh[0][2], wh[0][3],
                    wh[1][0], wh[1][1], wh[1][2], wh[1][3],
                    sel, acc[0], acc[1]);
#pragma unroll
    for (int q = 0; q < 2; q++) {
      const int j = jq[q] * 16 + col;
#pragma unroll
      for (int r = 0; r < 8; r++) {
        const int brow = mq[q] * 16 + r + 8 * sel;
        float iv = acc[q][0][r] + bj[q][0];
        float fv = acc[q][1][r] + bj[q][1];
        float gv = acc[q][2][r] + bj[q][2];
        float ov = acc[q][3][r] + bj[q][3];
        float cn = sigm(fv) * creg[q][r] + sigm(iv) * tanhf(gv);
        float hn = sigm(ov) * tanhf(cn);
        creg[q][r] = cn;
        hs[(size_t)brow * HID + j] = f2bf(hn);
        if (out_bf) {
          out_bf[((size_t)brow * SEQT + t) * (2 * HID) + (size_t)dir * HID + j] = f2bf(hn);
        } else {
          mreg[q][r] = fmaxf(mreg[q][r], hn);
          sreg[q][r] += hn;
        }
      }
    }
    ++epoch; dir_barrier(ctr, epoch * NWG);
  }

  if (out_pool) {
    const float invT = 1.0f / (float)SEQT;
#pragma unroll
    for (int q = 0; q < 2; q++) {
      const int j = jq[q] * 16 + col;
#pragma unroll
      for (int r = 0; r < 8; r++) {
        const int brow = mq[q] * 16 + r + 8 * sel;
        out_pool[(size_t)brow * (4 * HID) + (size_t)dir * HID + j]           = mreg[q][r];
        out_pool[(size_t)brow * (4 * HID) + 2 * HID + (size_t)dir * HID + j] = sreg[q][r] * invT;
      }
    }
  }
}

extern "C" void kernel_launch(void* const* d_in, const int* in_sizes, int n_in,
                              void* d_out, int out_size, void* d_ws, size_t ws_size,
                              hipStream_t stream) {
  (void)in_sizes; (void)n_in; (void)out_size; (void)ws_size;
  const int Bc = BATCH, Tc = SEQT, D1 = 128, D2 = 2 * HID;
  char* ws = (char*)d_ws;
  size_t cur = 0;
  auto take = [&](size_t bytes) -> size_t {
    size_t off = cur; cur = (cur + bytes + 255) & ~(size_t)255; return off;
  };
  const size_t off_ctr = take(4 * sizeof(unsigned));           // 4 barrier counters
  const size_t off_hs1 = take((size_t)2 * Bc * HID * 2);       // layer1 h state
  const size_t off_hs2 = take((size_t)2 * Bc * HID * 2);       // layer2 h state
  const size_t zero_bytes = cur;                               // zero everything above
  const size_t off_h2a = take((size_t)2 * Bc * HID * 2);
  const size_t off_h2b = take((size_t)2 * Bc * HID * 2);
  const size_t off_xt1 = take((size_t)2 * Bc * D1 * 2);
  const size_t off_xt2 = take((size_t)2 * Bc * D2 * 2);
  const size_t off_x   = take((size_t)Bc * Tc * D1 * 2);       // s as bf16
  const size_t off_r1  = take((size_t)Bc * Tc * D2 * 2);       // layer1 output bf16

  size_t off_w[4][4];
  const int szl1[4] = {GATES * D1, GATES * HID, D1 * HID, HID * D1};  // Wih,Whh,Q,R
  const int szl2[4] = {GATES * D2, GATES * HID, D2 * HID, HID * D2};
  for (int set = 0; set < 4; ++set) {
    const int* sz = (set < 2) ? szl1 : szl2;
    for (int w = 0; w < 4; ++w) off_w[set][w] = take((size_t)sz[w] * 2);
  }

  {
    int nwords = (int)(zero_bytes / 4);
    zero_u32_kernel<<<(nwords + 255) / 256, 256, 0, stream>>>((unsigned*)(ws + off_ctr), nwords);
  }

  auto conv = [&](const void* src, size_t off, int n) {
    f32_to_bf16_kernel<<<(n + 255) / 256, 256, 0, stream>>>(
        (const float*)src, (unsigned short*)(ws + off), n);
  };
  conv(d_in[0], off_x, Bc * Tc * D1);   // s -> bf16 (d_in[1]=lengths unused by reference)
  for (int set = 0; set < 4; ++set) {   // params order: l1_fwd,l1_bwd,l2_fwd,l2_bwd
    const int* sz = (set < 2) ? szl1 : szl2;
    const int base = 2 + set * 5;                      // Wih,Whh,b,Q,R per set
    const int map[4] = {base + 0, base + 1, base + 3, base + 4};
    for (int w = 0; w < 4; ++w) conv(d_in[map[w]], off_w[set][w], sz[w]);
  }

  auto Wp = [&](int s, int w) { return (const unsigned short*)(ws + off_w[s][w]); };

  // Layer 1: writes r1_bf [B,T,2H]
  mogrifier_bilstm_layer<128><<<dim3(2 * NWG), dim3(BLOCK), 0, stream>>>(
      (const unsigned short*)(ws + off_x),
      Wp(0, 0), Wp(0, 1), Wp(0, 2), Wp(0, 3), (const float*)d_in[2 + 0 * 5 + 2],
      Wp(1, 0), Wp(1, 1), Wp(1, 2), Wp(1, 3), (const float*)d_in[2 + 1 * 5 + 2],
      (unsigned short*)(ws + off_hs1), (unsigned short*)(ws + off_h2a),
      (unsigned short*)(ws + off_xt1),
      (unsigned*)(ws + off_ctr),
      (unsigned short*)(ws + off_r1), nullptr);

  // Layer 2: reads r1_bf, writes pooled [B,4H] f32 directly to d_out
  mogrifier_bilstm_layer<1024><<<dim3(2 * NWG), dim3(BLOCK), 0, stream>>>(
      (const unsigned short*)(ws + off_r1),
      Wp(2, 0), Wp(2, 1), Wp(2, 2), Wp(2, 3), (const float*)d_in[2 + 2 * 5 + 2],
      Wp(3, 0), Wp(3, 1), Wp(3, 2), Wp(3, 3), (const float*)d_in[2 + 3 * 5 + 2],
      (unsigned short*)(ws + off_hs2), (unsigned short*)(ws + off_h2b),
      (unsigned short*)(ws + off_xt2),
      (unsigned*)(ws + off_ctr) + 2,
      nullptr, (float*)d_out);
}
